// GCNBlock_79018808312518
// MI455X (gfx1250) — matrix operand order repaired and checked by reading, not compiled, not run on hardware
//
#include <hip/hip_runtime.h>
#include <hip/hip_bf16.h>

typedef __bf16 bf16;
typedef bf16  v16bf __attribute__((ext_vector_type(16)));
typedef bf16  v8bf  __attribute__((ext_vector_type(8)));
typedef float v8f   __attribute__((ext_vector_type(8)));
typedef unsigned int v4u __attribute__((ext_vector_type(4)));
typedef int   v8i   __attribute__((ext_vector_type(8)));
typedef int   v4i   __attribute__((ext_vector_type(4)));

#define B_      4
#define N_      16
#define T_      2048
#define F_      512
#define TT      8            // t's per workgroup
#define THREADS 256          // 8 wave32's

// -------- workspace layout (bf16 elements): weights pre-converted once --------
#define WS_W1   0
#define WS_W2   (512*1024)                  // 524288
#define WS_WM   (WS_W2 + 512*512)           // 786432
#define WS_EMB  (WS_WM + 512*512*16)        // 4980736
#define WS_TOT  (WS_EMB + 43*512)           // 5002752 bf16 elems (~9.6 MB)

// -------- dynamic LDS layout (bytes) --------
#define SM_H2   0                            // [8][512][16] bf16 = 131072 (h2 transposed: [slot][c][n])
#define SM_ND   131072                       // [16][1024]   bf16 = 32768  (node0)
#define SM_H1   163840                       // [16][512]    bf16 = 16384
#define SM_GT   180224                       // [8 waves][256] bf16 = 4096 (C->B re-layout scratch)
#define SM_EMA  184320                       // em_all [256][8] f32 = 8192 (TDM destination)
#define SM_BB   192512                       // [16][4]  f32 = 256
#define SM_CL   192768                       // [16] int = 64
#define SM_FT   SM_ND                        // feat [8][512] f32 = 16384, aliases node0 (post-loop only)
#define SMEM_BYTES 192832

__global__ __launch_bounds__(256) void prep_bf16(const float* __restrict__ W1,
                                                 const float* __restrict__ W2,
                                                 const float* __restrict__ Wm,
                                                 const float* __restrict__ emb,
                                                 bf16* __restrict__ ws) {
    int idx = blockIdx.x * 256 + threadIdx.x;
    if (idx < WS_W2)        ws[idx] = (bf16)W1[idx];
    else if (idx < WS_WM)   ws[idx] = (bf16)W2[idx - WS_W2];
    else if (idx < WS_EMB)  ws[idx] = (bf16)Wm[idx - WS_WM];
    else if (idx < WS_TOT)  ws[idx] = (bf16)emb[idx - WS_EMB];
}

__device__ __forceinline__ v16bf cat8(v8bf lo, v8bf hi) {
    return __builtin_shufflevector(lo, hi, 0,1,2,3,4,5,6,7,8,9,10,11,12,13,14,15);
}

__device__ __forceinline__ float relu1(float x) {          // single v_med3_f32
    return __builtin_amdgcn_fmed3f(x, 0.f, 3.0e38f);
}

__global__ __launch_bounds__(256, 1) void gcn_fused(
    const float* __restrict__ bbox, const int* __restrict__ bcls,
    const float* __restrict__ edge, const float* __restrict__ Wlin,
    const float* __restrict__ lnw,  const float* __restrict__ lnb,
    const bf16*  __restrict__ ws,   float* __restrict__ out) {

    extern __shared__ unsigned char smem[];
    bf16*  h2_s = (bf16*)(smem + SM_H2);
    bf16*  nd_s = (bf16*)(smem + SM_ND);
    bf16*  h1_s = (bf16*)(smem + SM_H1);
    bf16*  gt_s = (bf16*)(smem + SM_GT);
    float* ema_s= (float*)(smem + SM_EMA);
    float* bb_s = (float*)(smem + SM_BB);
    int*   cl_s = (int*)  (smem + SM_CL);
    float* ft_s = (float*)(smem + SM_FT);

    const int tid  = threadIdx.x;
    const int w    = tid >> 5;      // wave id (0..7)
    const int lane = tid & 31;
    const int g    = lane >> 4;     // lane group (A/B K-halves, k = 16*g + s in this code)
    const int ln   = lane & 15;
    const int b    = blockIdx.y;
    const int t0   = blockIdx.x * TT;

    const bf16* W1b = ws + WS_W1;
    const bf16* W2b = ws + WS_W2;
    const bf16* Wmb = ws + WS_WM;
    const bf16* Eb  = ws + WS_EMB;

    // ---------- TDM: fetch em[256 (i,j)][8 t's] for the whole t-group in one shot ----------
    // edge[b,i,j,t] : element (y = i*16+j, x = s) at base + 4*(y*2048 + x), y-stride 2048 elems.
    if (w == 0) {
        unsigned long long ga = (unsigned long long)(uintptr_t)edge
                              + 4ull * ((unsigned long long)b * (N_*N_*T_) + (unsigned)t0);
        v4u g0;
        g0[0] = 1u;                                        // count=1 (valid user descriptor)
        g0[1] = (unsigned)SM_EMA;                          // lds_addr (dynamic LDS base = 0)
        g0[2] = (unsigned)(ga & 0xffffffffu);              // global_addr[31:0]
        g0[3] = (unsigned)((ga >> 32) & 0x01ffffffu) | (2u << 30);  // addr[56:32] | type=2
        v8i g1;
        g1[0] = 0x20000;                                   // data_size=2 (4 bytes)
        g1[1] = (int)(8u   << 16);                         // tensor_dim0 = 8  (bits 79:48)
        g1[2] = (int)(256u << 16);                         // tensor_dim1 = 256 (bits 111:80)
        g1[3] = (int)(8u   << 16);                         // tile_dim0 = 8    (bits 127:112)
        g1[4] = 256;                                       // tile_dim1 = 256, tile_dim2 = 0
        g1[5] = 2048;                                      // tensor_dim0_stride = 2048 elems
        g1[6] = 0;
        g1[7] = 0;
        v4i gz4 = {0, 0, 0, 0};
        v8i gz8 = {0, 0, 0, 0, 0, 0, 0, 0};
        __builtin_amdgcn_tensor_load_to_lds(g0, g1, gz4, gz4, gz8, 0);
        __builtin_amdgcn_s_wait_tensorcnt(0);
    }

    for (int s = 0; s < TT; ++s) {
        const int t = t0 + s;
        // ---------- phase 0a: stage per-t scalars ----------
        if (tid < N_)    cl_s[tid] = bcls[((size_t)b*N_ + tid)*T_ + t];
        if (tid < N_*4){ int n = tid >> 2, c = tid & 3;
            bb_s[tid] = bbox[(((size_t)b*N_ + n)*4 + c)*T_ + t]; }
        __syncthreads();   // also orders TDM em tile (wave0 waited tensorcnt) for all waves

        // em as WMMA A operand (16x32, K=16..31 zero-padded)
        v16bf aem;
        #pragma unroll
        for (int h = 0; h < 16; ++h) aem[h] = (g == 0) ? (bf16)ema_s[(ln*16 + h)*8 + s] : (bf16)0.f;

        // ---------- phase 0b: build node0 [16][1024] bf16 = (emb || bbox@Wlin^T) ----------
        {   int n = tid >> 4, c0 = (tid & 15) * 64;
            if (c0 < 512) {
                const bf16* src = Eb + (size_t)cl_s[n]*F_ + c0;
                bf16* dst = nd_s + n*1024 + c0;
                #pragma unroll 8
                for (int j = 0; j < 64; ++j) dst[j] = src[j];
            } else {
                float b0 = bb_s[n*4+0], b1 = bb_s[n*4+1], b2 = bb_s[n*4+2], b3 = bb_s[n*4+3];
                bf16* dst = nd_s + n*1024 + c0;
                const float* wl = Wlin + (size_t)(c0 - 512)*4;
                #pragma unroll 8
                for (int j = 0; j < 64; ++j)
                    dst[j] = (bf16)(b0*wl[4*j] + b1*wl[4*j+1] + b2*wl[4*j+2] + b3*wl[4*j+3]);
            }
        }
        __syncthreads();

        // ---------- phase 1: h1 = relu(em @ (node0 @ W1^T)), per-wave F-tiles ----------
        for (int ft = 0; ft < 4; ++ft) {
            const int fb = (w*4 + ft) * 16;
            v8f c = {};
            #pragma unroll 4
            for (int kt = 0; kt < 32; ++kt) {
                const int k0 = kt*32 + 16*g;
                v8bf lo = *(const v8bf*)(nd_s + ln*1024 + k0);
                v8bf hi = *(const v8bf*)(nd_s + ln*1024 + k0 + 8);
                v16bf a  = cat8(lo, hi);
                v16bf bm = *(const v16bf*)(W1b + (size_t)(fb + ln)*1024 + kt*32 + 16*g);
                c = __builtin_amdgcn_wmma_f32_16x16x32_bf16(false, a, false, bm, (short)0, c, false, false);
            }
            // C (f32) -> bf16 B operand via per-wave LDS scratch, stored transposed
            {   v8bf gv;
                #pragma unroll
                for (int r = 0; r < 8; ++r) gv[r] = (bf16)c[r];
                *(v8bf*)(gt_s + w*256 + ln*16 + 8*g) = gv; }
            asm volatile("s_wait_dscnt 0" ::: "memory");
            v16bf bg;
            if (g == 0) bg = *(const v16bf*)(gt_s + w*256 + ln*16);
            else {
                #pragma unroll
                for (int h = 0; h < 16; ++h) bg[h] = (bf16)0.f;
            }
            v8f c2 = {};
            c2 = __builtin_amdgcn_wmma_f32_16x16x32_bf16(false, aem, false, bg, (short)0, c2, false, false);
            #pragma unroll
            for (int r = 0; r < 8; ++r)
                h1_s[(r + 8*g)*F_ + fb + ln] = (bf16)relu1(c2[r]);
        }
        __syncthreads();

        // ---------- phase 2: h2 = relu(em @ (h1 @ W2^T)); store transposed [slot][c][n] ----------
        for (int ft = 0; ft < 4; ++ft) {
            const int fb = (w*4 + ft) * 16;
            v8f c = {};
            #pragma unroll 4
            for (int kt = 0; kt < 16; ++kt) {
                const int k0 = kt*32 + 16*g;
                v8bf lo = *(const v8bf*)(h1_s + ln*F_ + k0);
                v8bf hi = *(const v8bf*)(h1_s + ln*F_ + k0 + 8);
                v16bf a  = cat8(lo, hi);
                v16bf bm = *(const v16bf*)(W2b + (size_t)(fb + ln)*F_ + kt*32 + 16*g);
                c = __builtin_amdgcn_wmma_f32_16x16x32_bf16(false, a, false, bm, (short)0, c, false, false);
            }
            {   v8bf gv;
                #pragma unroll
                for (int r = 0; r < 8; ++r) gv[r] = (bf16)c[r];
                *(v8bf*)(gt_s + w*256 + ln*16 + 8*g) = gv; }
            asm volatile("s_wait_dscnt 0" ::: "memory");
            v16bf bg;
            if (g == 0) bg = *(const v16bf*)(gt_s + w*256 + ln*16);
            else {
                #pragma unroll
                for (int h = 0; h < 16; ++h) bg[h] = (bf16)0.f;
            }
            v8f c2 = {};
            c2 = __builtin_amdgcn_wmma_f32_16x16x32_bf16(false, aem, false, bg, (short)0, c2, false, false);
            v8bf hv;
            #pragma unroll
            for (int r = 0; r < 8; ++r) hv[r] = (bf16)relu1(c2[r]);
            // h2t[s][c=fb+ln][n=8g..8g+7] : one aligned 16B chunk per lane
            *(v8bf*)(h2_s + ((size_t)s*F_ + fb + ln)*N_ + 8*g) = hv;
        }
        __syncthreads();
    }

    // ---------- phase 3: feat[s][f] = sum_{c,n} h2[s,n,c]*Wm[f,c,n] as M=16 WMMA GEMM, K=8192 ----------
    for (int ft = 0; ft < 4; ++ft) {
        const int fb = (w*4 + ft) * 16;
        const int m8 = ln & 7;                       // rows 8..15 duplicate slots 0..7 (discarded)
        v8f c = {};
        #pragma unroll 4
        for (int kt = 0; kt < 256; ++kt) {
            const int k0 = kt*32 + 16*g;             // k = c*16 + n, h2_s is k-linear per slot
            v8bf lo = *(const v8bf*)(h2_s + (size_t)m8*(F_*N_) + k0);
            v8bf hi = *(const v8bf*)(h2_s + (size_t)m8*(F_*N_) + k0 + 8);
            v16bf a  = cat8(lo, hi);
            v16bf bm = *(const v16bf*)(Wmb + (size_t)(fb + ln)*(F_*N_) + kt*32 + 16*g);
            c = __builtin_amdgcn_wmma_f32_16x16x32_bf16(false, a, false, bm, (short)0, c, false, false);
        }
        if (g == 0) {
            #pragma unroll
            for (int r = 0; r < 8; ++r) ft_s[r*F_ + fb + ln] = c[r];
        }
    }
    __syncthreads();

    // ---------- phase 4: LayerNorm over F per slot (wave w owns slot w) ----------
    {
        float sum = 0.f, sq = 0.f, vals[16];
        #pragma unroll
        for (int i = 0; i < 16; ++i) {
            float v = ft_s[w*F_ + lane + 32*i];
            vals[i] = v; sum += v; sq += v*v;
        }
        #pragma unroll
        for (int off = 16; off > 0; off >>= 1) {
            sum += __shfl_xor(sum, off, 32);
            sq  += __shfl_xor(sq,  off, 32);
        }
        const float mu  = sum * (1.f/F_);
        const float var = sq * (1.f/F_) - mu*mu;
        const float inv = rsqrtf(var + 1e-5f);
        #pragma unroll
        for (int i = 0; i < 16; ++i) {
            int f = lane + 32*i;
            ft_s[w*F_ + f] = (vals[i] - mu) * inv * lnw[f] + lnb[f];
        }
    }
    __syncthreads();

    // ---------- output [B,F,T]: 8 consecutive t's per f -> two aligned float4 stores ----------
    for (int f = tid; f < F_; f += THREADS) {
        float4 o0, o1;
        o0.x = ft_s[0*F_+f]; o0.y = ft_s[1*F_+f]; o0.z = ft_s[2*F_+f]; o0.w = ft_s[3*F_+f];
        o1.x = ft_s[4*F_+f]; o1.y = ft_s[5*F_+f]; o1.z = ft_s[6*F_+f]; o1.w = ft_s[7*F_+f];
        float* dst = out + ((size_t)b*F_ + f)*T_ + t0;
        *(float4*)(dst)     = o0;
        *(float4*)(dst + 4) = o1;
    }
}

extern "C" void kernel_launch(void* const* d_in, const int* in_sizes, int n_in,
                              void* d_out, int out_size, void* d_ws, size_t ws_size,
                              hipStream_t stream) {
    (void)in_sizes; (void)n_in; (void)out_size; (void)ws_size;
    const float* bbox = (const float*)d_in[0];
    const int*   bcls = (const int*)  d_in[1];
    const float* edge = (const float*)d_in[2];
    const float* Wlin = (const float*)d_in[3];
    const float* emb  = (const float*)d_in[4];
    const float* W1   = (const float*)d_in[5];
    const float* W2   = (const float*)d_in[6];
    const float* Wm   = (const float*)d_in[7];
    const float* lnw  = (const float*)d_in[8];
    const float* lnb  = (const float*)d_in[9];
    bf16*  ws  = (bf16*)d_ws;
    float* out = (float*)d_out;

    (void)hipFuncSetAttribute((const void*)gcn_fused,
                              hipFuncAttributeMaxDynamicSharedMemorySize, SMEM_BYTES);

    prep_bf16<<<(WS_TOT + 255) / 256, 256, 0, stream>>>(W1, W2, Wm, emb, ws);

    dim3 grid(T_ / TT, B_);
    gcn_fused<<<grid, THREADS, SMEM_BYTES, stream>>>(bbox, bcls, edge, Wlin, lnw, lnb, ws, out);
}
